// AdaMoLE_44822278701275
// MI455X (gfx1250) — matrix operand order repaired and checked by reading, not compiled, not run on hardware
//
#include <hip/hip_runtime.h>
#include <hip/hip_bf16.h>

// ---------------------------------------------------------------------------
// AdaMoLE fused kernel for MI455X (gfx1250, wave32, WMMA bf16 K=32).
//
// N=16384 tokens, D_IN=D_OUT=4096, E=8, R=16, K2=E*R=128.
//   pack kernels : (Wr||Wt, A) -> per-K-chunk contiguous bf16 "packages",
//                  Bw -> bf16 WMMA B-layout tiles, all in d_ws.
//   main kernel  : 1 wave per 16-token tile, 4 waves/block.
//     stage 1: weight packages stream into double-buffered LDS via
//              global_load_async_to_lds_b128 (ASYNCcnt), shared by all 4
//              waves; x streams global->reg->bf16 with ping/pong register
//              sets (2x manual unroll, no WAR rotation); 9 WMMAs per chunk.
//     epilogue: softmax + sigmoid threshold + weight normalize (LDS).
//     stage 2: (h*w)[16x128] @ Bw[128x4096], 4 WMMAs per 16-col tile,
//              register double-buffered B fragments.
// ---------------------------------------------------------------------------

typedef __attribute__((ext_vector_type(16))) __bf16 v16bf;
typedef __attribute__((ext_vector_type(8)))  float  v8f;
typedef __attribute__((ext_vector_type(4)))  float  v4f;

#define D_IN    4096
#define D_OUT   4096
#define N_TOK   16384
#define N_EXP   8
#define RANK    16
#define SCALING 0.5f            // lora_alpha / r = 8/16
#define KT1     (D_IN / 32)     // 128 K-chunks for stage 1
#define K2      (N_EXP * RANK)  // 128
#define KT2     (K2 / 32)       // 4 K-chunks for stage 2
#define NT2     (D_OUT / 16)    // 256 output column tiles
#define WAVES_PER_BLOCK 4
#define NFRAG   (N_EXP + 1)     // 9 fragments per K-chunk (router + 8 experts)

// one weight package = 9 frags * 32 lanes * 16 bf16 = 9216 B, padded to
// 10240 B so 128 threads each issue exactly 5 async B128 loads per package.
#define PKG_PAD_ELEMS  5120                     // 10240 bytes
#define PKG_PAD_BYTES  (PKG_PAD_ELEMS * 2)
#define ASYNC_PER_THREAD 5                      // 5 * 128 * 16B = 10240 B

// workspace layout (bf16 elements)
#define PKG_OFF    0
#define PKG_ELEMS  (KT1 * PKG_PAD_ELEMS)        // 655360
#define BWPACK_OFF (PKG_OFF + PKG_ELEMS)
#define BWPACK_ELEMS (NT2 * KT2 * 32 * 16)      // 524288

// B-matrix per-lane layout assumed (16-bit, 32x16 tile):
//   lane L holds column N = L&15; K elements (L>>4)*8 + 0..7, 16 + (L>>4)*8 + 0..7, 2 per VGPR.
// Pack kernels store 16 contiguous bf16 per (tile, lane).

__global__ void adamole_pack_pkg(const float* __restrict__ Wr,
                                 const float* __restrict__ Wt,
                                 const float* __restrict__ A,
                                 __bf16* __restrict__ ws) {
    int tid = blockIdx.x * blockDim.x + threadIdx.x;   // KT1*NFRAG*32 threads
    if (tid >= KT1 * NFRAG * 32) return;
    int t    = tid / (NFRAG * 32);
    int rem  = tid - t * (NFRAG * 32);
    int f    = rem >> 5;                   // fragment: 0 = router||thr, 1+e = A[e]
    int lane = rem & 31;
    int col  = lane & 15;
    int kb   = t * 32 + (lane >> 4) * 8;
    __bf16* dst = ws + PKG_OFF + (size_t)t * PKG_PAD_ELEMS + ((size_t)f * 32 + lane) * 16;
#pragma unroll
    for (int i = 0; i < 16; ++i) {
        int k = kb + (i & 7) + ((i >> 3) << 4);
        float v = 0.0f;
        if (f == 0) {
            if (col < N_EXP)       v = Wr[(size_t)col * D_IN + k];
            else if (col == N_EXP) v = Wt[k];
        } else {
            int e = f - 1;
            v = A[((size_t)e * D_IN + k) * RANK + col];   // A[e][k][r=col]
        }
        dst[i] = (__bf16)v;
    }
}

__global__ void adamole_pack_bw(const float* __restrict__ Bw,
                                __bf16* __restrict__ ws) {
    int tid = blockIdx.x * blockDim.x + threadIdx.x;   // NT2*KT2*32 threads
    if (tid >= NT2 * KT2 * 32) return;
    int nt   = tid >> 7;
    int rem  = tid & (KT2 * 32 - 1);
    int kt   = rem >> 5;
    int lane = rem & 31;
    int d    = nt * 16 + (lane & 15);
    int kb   = kt * 32 + (lane >> 4) * 8;
    __bf16* dst = ws + BWPACK_OFF + (size_t)tid * 16;
#pragma unroll
    for (int i = 0; i < 16; ++i) {
        // Bw flattened: [(e*R + r), D_OUT] == [K2, D_OUT]
        dst[i] = (__bf16)Bw[(size_t)(kb + (i & 7) + ((i >> 3) << 4)) * D_OUT + d];
    }
}

__global__ __launch_bounds__(WAVES_PER_BLOCK * 32)
void adamole_main(const float* __restrict__ x,
                  const float* __restrict__ br,
                  const float* __restrict__ bt,
                  const __bf16* __restrict__ ws,
                  float* __restrict__ out) {
    // double-buffered weight-package staging (filled by async-to-LDS)
    __shared__ __align__(16) __bf16 wbuf[2][PKG_PAD_ELEMS];
    // wave-private staging (padded rows to dodge bank conflicts)
    __shared__ float lds_h[WAVES_PER_BLOCK][16][K2 + 4];
    __shared__ float lds_lg[WAVES_PER_BLOCK][16][17];
    __shared__ float lds_w[WAVES_PER_BLOCK][16][N_EXP];

    const int tid   = threadIdx.x;
    const int lane  = tid & 31;
    const int wave  = tid >> 5;
    const int tile  = blockIdx.x * WAVES_PER_BLOCK + wave;  // 16-token tile
    const int row0  = tile * 16;
    const int m     = lane & 15;       // token row (A-frag) / column (B/C)
    const int khalf = lane >> 4;

    const __bf16* pkg = ws + PKG_OFF;
    const __bf16* Bwp = ws + BWPACK_OFF;
    const unsigned lds0 = (unsigned)(size_t)(void*)&wbuf[0][0];

    // issue one weight package (10240 B) into LDS buffer `buf` (async copy)
    auto issue_pkg = [&](int t, int buf) {
        const __bf16* src = pkg + (size_t)t * PKG_PAD_ELEMS;
        const unsigned dbase = lds0 + (unsigned)buf * PKG_PAD_BYTES;
#pragma unroll
        for (int j = 0; j < ASYNC_PER_THREAD; ++j) {
            const int idx = tid + j * 128;          // 16-byte chunk index
            asm volatile("global_load_async_to_lds_b128 %0, %1, off"
                         :
                         : "v"(dbase + (unsigned)idx * 16), "v"(src + (size_t)idx * 8)
                         : "memory");
        }
    };

    // ---------------- stage 1: logits + per-expert LoRA down-proj ----------
    v8f accL = {};
    v8f accH[N_EXP];
#pragma unroll
    for (int e = 0; e < N_EXP; ++e) accH[e] = (v8f){};

    const float* xrow = x + (size_t)(row0 + m) * D_IN;

    // one half-iteration: consume chunk c (x floats in x0..x3, weights in
    // wbuf[c&1]); issue package c+1; reload x0..x3 with chunk c+2.
    auto half = [&](int c, v4f& x0, v4f& x1, v4f& x2, v4f& x3) {
        const int nxt = (c + 1 < KT1) ? (c + 1) : c;     // dummy reissue at end
        issue_pkg(nxt, (c + 1) & 1);

        // package c complete once only package c+1's 5 loads remain
        asm volatile("s_wait_asynccnt 0x5" ::: "memory");
        __syncthreads();

        // build 16x32 bf16 A-fragment (lane holds token row m):
        // VGPR0-3 <- K = 8*khalf + 0..7 ; VGPR4-7 <- K = 16 + 8*khalf + 0..7
        v16bf a;
#pragma unroll
        for (int i = 0; i < 4; ++i) {
            a[i]      = (__bf16)x0[i];
            a[4 + i]  = (__bf16)x1[i];
            a[8 + i]  = (__bf16)x2[i];
            a[12 + i] = (__bf16)x3[i];
        }

        // load all 9 B-fragments from LDS, then issue 9 WMMAs
        const v16bf* fb = (const v16bf*)&wbuf[c & 1][0];
        v16bf bW = fb[lane];
        v16bf bA[N_EXP];
#pragma unroll
        for (int e = 0; e < N_EXP; ++e) bA[e] = fb[(e + 1) * 32 + lane];

        accL = __builtin_amdgcn_wmma_f32_16x16x32_bf16(
            false, a, false, bW, (short)0, accL, false, false);
#pragma unroll
        for (int e = 0; e < N_EXP; ++e) {
            accH[e] = __builtin_amdgcn_wmma_f32_16x16x32_bf16(
                false, a, false, bA[e], (short)0, accH[e], false, false);
        }

        // reload this register set with chunk c+2 (clamped; issued after use)
        const int pc = (c + 2 < KT1) ? (c + 2) : (KT1 - 1);
        const float* p0 = xrow + pc * 32 + 8 * khalf;
        x0 = *(const v4f*)(p0);
        x1 = *(const v4f*)(p0 + 4);
        x2 = *(const v4f*)(p0 + 16);
        x3 = *(const v4f*)(p0 + 20);

        __syncthreads();   // all waves done reading wbuf[c&1] before overwrite
    };

    // prologue: package 0 in flight; x chunks 0 (set A) and 1 (set B)
    issue_pkg(0, 0);
    v4f xa0, xa1, xa2, xa3, xb0, xb1, xb2, xb3;
    {
        const float* p0 = xrow + 8 * khalf;
        xa0 = *(const v4f*)(p0);
        xa1 = *(const v4f*)(p0 + 4);
        xa2 = *(const v4f*)(p0 + 16);
        xa3 = *(const v4f*)(p0 + 20);
        const float* p1 = xrow + 32 + 8 * khalf;
        xb0 = *(const v4f*)(p1);
        xb1 = *(const v4f*)(p1 + 4);
        xb2 = *(const v4f*)(p1 + 16);
        xb3 = *(const v4f*)(p1 + 20);
    }

    for (int t = 0; t < KT1; t += 2) {
        half(t,     xa0, xa1, xa2, xa3);
        half(t + 1, xb0, xb1, xb2, xb3);
    }

    // spill C-layout accumulators: VGPR j <-> (tok = j + 8*khalf, col = m)
#pragma unroll
    for (int j = 0; j < 8; ++j) {
        lds_lg[wave][j + 8 * khalf][m] = accL[j];
    }
#pragma unroll
    for (int e = 0; e < N_EXP; ++e) {
#pragma unroll
        for (int j = 0; j < 8; ++j) {
            lds_h[wave][j + 8 * khalf][e * 16 + m] = accH[e][j];
        }
    }
    __syncthreads();

    // ---------------- epilogue: softmax, threshold, weight normalize -------
    if (lane < 16) {
        const int tok = lane;
        float l[N_EXP];
        float mx = -3.0e38f;
#pragma unroll
        for (int e = 0; e < N_EXP; ++e) {
            l[e] = lds_lg[wave][tok][e] + br[e];
            mx = fmaxf(mx, l[e]);
        }
        float tl = lds_lg[wave][tok][N_EXP] + bt[0];
        float se = 0.0f;
#pragma unroll
        for (int e = 0; e < N_EXP; ++e) { l[e] = __expf(l[e] - mx); se += l[e]; }
        const float inv_se = 1.0f / se;
        const float thr = (1.0f / (1.0f + __expf(-tl))) * (1.0f / (float)N_EXP);
        float w[N_EXP];
        float sw = 0.0f;
#pragma unroll
        for (int e = 0; e < N_EXP; ++e) {
            float ad = l[e] * inv_se - thr;
            w[e] = (ad >= 0.0f) ? ad : 0.0f;
            sw += w[e];
        }
        if (sw == 0.0f) sw = 1.0f;
        const float scl = SCALING / sw;
#pragma unroll
        for (int e = 0; e < N_EXP; ++e) lds_w[wave][tok][e] = w[e] * scl;
    }
    __syncthreads();

    // ---------------- stage 2: (h*w) [16x128] @ Bw [128 x D_OUT] -----------
    // build the 4 A-fragments once, folding routing weight + scaling
    v16bf afrag[KT2];
#pragma unroll
    for (int kt = 0; kt < KT2; ++kt) {
        const int kb0 = kt * 32 + 8 * khalf;
        const int kb1 = kb0 + 16;
#pragma unroll
        for (int i = 0; i < 8; ++i) {
            const int ka = kb0 + i, kb = kb1 + i;
            afrag[kt][i]     = (__bf16)(lds_h[wave][m][ka] * lds_w[wave][m][ka >> 4]);
            afrag[kt][8 + i] = (__bf16)(lds_h[wave][m][kb] * lds_w[wave][m][kb >> 4]);
        }
    }

    // register double-buffered B fragments across the 256 output tiles
    v16bf bcur[KT2], bnxt[KT2];
#pragma unroll
    for (int kt = 0; kt < KT2; ++kt)
        bcur[kt] = *(const v16bf*)(Bwp + ((size_t)kt * 32 + lane) * 16);

    for (int nt = 0; nt < NT2; ++nt) {
        const int nn = (nt + 1 < NT2) ? (nt + 1) : nt;
#pragma unroll
        for (int kt = 0; kt < KT2; ++kt)
            bnxt[kt] = *(const v16bf*)(Bwp + (((size_t)nn * KT2 + kt) * 32 + lane) * 16);

        v8f acc = {};
#pragma unroll
        for (int kt = 0; kt < KT2; ++kt) {
            acc = __builtin_amdgcn_wmma_f32_16x16x32_bf16(
                false, afrag[kt], false, bcur[kt], (short)0, acc, false, false);
        }
        // store C: VGPR j -> out[row0 + j + 8*khalf][nt*16 + m]
#pragma unroll
        for (int j = 0; j < 8; ++j) {
            out[(size_t)(row0 + j + 8 * khalf) * D_OUT + nt * 16 + m] = acc[j];
        }
#pragma unroll
        for (int kt = 0; kt < KT2; ++kt) bcur[kt] = bnxt[kt];
    }
}

// ---------------------------------------------------------------------------
extern "C" void kernel_launch(void* const* d_in, const int* in_sizes, int n_in,
                              void* d_out, int out_size, void* d_ws, size_t ws_size,
                              hipStream_t stream) {
    (void)in_sizes; (void)n_in; (void)out_size; (void)ws_size;
    const float* x  = (const float*)d_in[0];   // [4,4096,4096]
    const float* Wr = (const float*)d_in[1];   // [8,4096]
    const float* br = (const float*)d_in[2];   // [8]
    const float* Wt = (const float*)d_in[3];   // [1,4096]
    const float* bt = (const float*)d_in[4];   // [1]
    const float* A  = (const float*)d_in[5];   // [8,4096,16]
    const float* Bw = (const float*)d_in[6];   // [8,16,4096]
    float* out = (float*)d_out;                // [4,4096,4096]
    __bf16* ws = (__bf16*)d_ws;                // needs ~2.4 MB

    // pack weights to bf16 (stream-ordered before main kernel)
    adamole_pack_pkg<<<(KT1 * NFRAG * 32 + 255) / 256, 256, 0, stream>>>(Wr, Wt, A, ws);
    adamole_pack_bw<<<(NT2 * KT2 * 32 + 255) / 256, 256, 0, stream>>>(Bw, ws);

    // 1024 token tiles, 4 waves (tiles) per 128-thread block
    const int n_tiles = N_TOK / 16;
    adamole_main<<<n_tiles / WAVES_PER_BLOCK, WAVES_PER_BLOCK * 32, 0, stream>>>(
        x, br, bt, ws, out);
}
